// FeaturewiseODERNNEncoder_40776419508622
// MI455X (gfx1250) — matrix operand order repaired and checked by reading, not compiled, not run on hardware
//
#include <hip/hip_runtime.h>
#include <hip/hip_bf16.h>

typedef __attribute__((ext_vector_type(16))) _Float16 v16h;
typedef __attribute__((ext_vector_type(8)))  _Float16 v8h;
typedef __attribute__((ext_vector_type(8)))  float    v8f;

#define T_STEPS 1024
#define NFEAT   2048
#define LAT     128
#define ODE_H   256
#define WAVES   2

// Weight fragment buffer layout (halfs):
//   W1t  (K=128 -> N=256): tiles = ntile*4 + ktile, 16*4 tiles   @ offset 0
//   W2t  (K=256 -> N=128): tiles = ntile*8 + ktile,  8*8 tiles   @ offset 32768
//   Whht (K=128 -> N=384): tiles = ntile*4 + ktile, 24*4 tiles   @ offset 65536
#define W1T_OFF  0
#define W2T_OFF  32768
#define WHT_OFF  65536
#define WHALFS   114688
#define WBYTES   (WHALFS * 2)          // 229376 bytes of f16 weight fragments

// Dynamic LDS carve (bytes), all 16B aligned:
#define LDS_W     0
#define LDS_H32   (WBYTES)                                   // WAVES*16*128*4
#define LDS_H16   (LDS_H32 + WAVES * 16 * LAT * 4)           // WAVES*16*128*2
#define LDS_ACT   (LDS_H16 + WAVES * 16 * LAT * 2)           // WAVES*16*256*2
#define LDS_OBS   (LDS_ACT + WAVES * 16 * ODE_H * 2)
#define LDS_MSK   (LDS_OBS + WAVES * 16 * 4)
#define LDS_TOTAL (LDS_MSK + WAVES * 16 * 4)                 // = 270592

__device__ __forceinline__ float fast_sigmoid(float x) {
  return 1.0f / (1.0f + __expf(-x));
}

__device__ __forceinline__ float fast_tanh(float x) {
#if __has_builtin(__builtin_amdgcn_tanhf)
  return __builtin_amdgcn_tanhf(x);
#else
  float e = __expf(2.0f * x);
  return (e - 1.0f) / (e + 1.0f);
#endif
}

__device__ __forceinline__ void wait_asynccnt0() {
#if __has_builtin(__builtin_amdgcn_s_wait_asynccnt)
  __builtin_amdgcn_s_wait_asynccnt(0);
#else
  asm volatile("s_wait_asynccnt 0" ::: "memory");
#endif
}

__device__ __forceinline__ v8f wmma_f16(v16h a, v16h b, v8f c) {
  return __builtin_amdgcn_wmma_f32_16x16x32_f16(false, a, false, b, (short)0, c,
                                                false, false);
}

// 16x32 f16 A-fragment for row M=ln from a K-contiguous f16 row in LDS.
__device__ __forceinline__ v16h load_afrag(const _Float16* row, int kbase, int g) {
  v8h lo = *(const v8h*)(row + kbase + g * 16);
  v8h hi = *(const v8h*)(row + kbase + g * 16 + 8);
  return __builtin_shufflevector(lo, hi, 0, 1, 2, 3, 4, 5, 6, 7,
                                 8, 9, 10, 11, 12, 13, 14, 15);
}

// ---------------------------------------------------------------------------
// Prep kernel: f32 weights -> f16 B-fragment layout in d_ws.
//   tile = (n/16)*KT + (k/32); lane = (n%16) + 16*((k%32)>>4); e = k%16
//   offset = tile*512 + lane*16 + e (halfs; lane's 16 halfs contiguous)
// ---------------------------------------------------------------------------
__global__ void prep_weights_kernel(const float* __restrict__ W1,
                                    const float* __restrict__ W2,
                                    const float* __restrict__ Whh,
                                    _Float16* __restrict__ wf) {
  int t = blockIdx.x * blockDim.x + threadIdx.x;
  if (t < 32768) {
    int n = t >> 7, k = t & 127;
    int tile = (n >> 4) * 4 + (k >> 5);
    int lane = (n & 15) + (((k >> 4) & 1) << 4);
    wf[W1T_OFF + tile * 512 + lane * 16 + (k & 15)] = (_Float16)W1[n * 128 + k];
  } else if (t < 65536) {
    int t2 = t - 32768;
    int k = t2 >> 7, n = t2 & 127;
    int tile = (n >> 4) * 8 + (k >> 5);
    int lane = (n & 15) + (((k >> 4) & 1) << 4);
    wf[W2T_OFF + tile * 512 + lane * 16 + (k & 15)] = (_Float16)W2[n * 256 + k];
  } else if (t < WHALFS) {
    int t3 = t - 65536;
    int k = t3 / 384, n = t3 % 384;
    int tile = (n >> 4) * 4 + (k >> 5);
    int lane = (n & 15) + (((k >> 4) & 1) << 4);
    wf[WHT_OFF + tile * 512 + lane * 16 + (k & 15)] = (_Float16)Whh[n * 128 + k];
  }
}

// ---------------------------------------------------------------------------
// Main kernel: one wave owns a 16-feature slab of h, loops over T.
// Weights live in LDS (staged once via async global->LDS b128 copies).
// ---------------------------------------------------------------------------
__global__ __launch_bounds__(WAVES * 32)
void ode_rnn_encoder_kernel(const float* __restrict__ times,
                            const float* __restrict__ vals,
                            const float* __restrict__ mask,
                            const float* __restrict__ b1,
                            const float* __restrict__ b2,
                            const float* __restrict__ W_ih,
                            const float* __restrict__ b_ih,
                            const float* __restrict__ b_hh,
                            const _Float16* __restrict__ wf,
                            float* __restrict__ out) {
  extern __shared__ char smem[];
  _Float16* wlds = (_Float16*)(smem + LDS_W);

  const int tid  = threadIdx.x;
  const int wid  = tid >> 5;
  const int lane = tid & 31;
  const int g    = lane >> 4;   // lane group (0/1)
  const int ln   = lane & 15;   // N-index for B/C/D, M-row for A
  const int f0   = blockIdx.x * (WAVES * 16) + wid * 16;

  float*    h32w = (float*)(smem + LDS_H32) + wid * 16 * LAT;
  _Float16* h16w = (_Float16*)(smem + LDS_H16) + wid * 16 * LAT;
  _Float16* actw = (_Float16*)(smem + LDS_ACT) + wid * 16 * ODE_H;
  float*    obsw = (float*)(smem + LDS_OBS) + wid * 16;
  float*    mskw = (float*)(smem + LDS_MSK) + wid * 16;

  // ---- Stage all weight fragments into LDS with async global->LDS copies.
  {
    const uint32_t lds_base = (uint32_t)(uintptr_t)(void*)wlds;  // LDS offset
    const char* gbase = (const char*)wf;
    for (int idx = tid; idx < WBYTES / 16; idx += WAVES * 32) {
      uint32_t loff = lds_base + (uint32_t)idx * 16u;
      const void* ga = gbase + (size_t)idx * 16u;
      asm volatile("global_load_async_to_lds_b128 %0, %1, off"
                   :: "v"(loff), "v"(ga)
                   : "memory");
    }
    wait_asynccnt0();
  }

  const _Float16* W1t = wlds + W1T_OFF;
  const _Float16* W2t = wlds + W2T_OFF;
  const _Float16* Wht = wlds + WHT_OFF;

  // h0 = 0
  for (int i = lane; i < 16 * LAT; i += 32) {
    h32w[i] = 0.0f;
    h16w[i] = (_Float16)0.0f;
  }
  __syncthreads();   // weights staged + state initialized

  // ---- Hoist per-lane bias / W_ih constants into registers.
  float b1r[16];
#pragma unroll
  for (int nt = 0; nt < 16; ++nt) b1r[nt] = b1[nt * 16 + ln];
  float b2r[8];
#pragma unroll
  for (int nt = 0; nt < 8; ++nt) b2r[nt] = b2[nt * 16 + ln];
  float wir[8], wiz[8], win[8], bir[8], biz[8], bin[8], bhr[8], bhz[8], bhn[8];
#pragma unroll
  for (int nt = 0; nt < 8; ++nt) {
    const int n = nt * 16 + ln;
    wir[nt] = W_ih[n];       wiz[nt] = W_ih[128 + n];  win[nt] = W_ih[256 + n];
    bir[nt] = b_ih[n];       biz[nt] = b_ih[128 + n];  bin[nt] = b_ih[256 + n];
    bhr[nt] = b_hh[n];       bhz[nt] = b_hh[128 + n];  bhn[nt] = b_hh[256 + n];
  }

  for (int t = 0; t < T_STEPS; ++t) {
    const int ri = T_STEPS - 1 - t;                  // original time index
    const float dt = (t == 0) ? 0.0f : (times[ri + 1] - times[ri]);

    if (lane < 16) {
      obsw[lane] = vals[ri * NFEAT + f0 + lane];
      mskw[lane] = mask[ri * NFEAT + f0 + lane];
    }

    // ---------------- ODE Euler update (uniform branch on scalar dt) -------
    if (dt > 0.0f) {
      v16h ah[4];
#pragma unroll
      for (int kt = 0; kt < 4; ++kt)
        ah[kt] = load_afrag(h16w + ln * LAT, kt * 32, g);

      // GEMM1: (16x128) x (128x256) -> tanh -> act
#pragma unroll
      for (int nt = 0; nt < 16; ++nt) {
        v8f c = {};
#pragma unroll
        for (int kt = 0; kt < 4; ++kt) {
          v16h b = *(const v16h*)(W1t + (nt * 4 + kt) * 512 + lane * 16);
          c = wmma_f16(ah[kt], b, c);
        }
        const int n = nt * 16 + ln;
#pragma unroll
        for (int r = 0; r < 8; ++r)
          actw[(r + 8 * g) * ODE_H + n] = (_Float16)fast_tanh(c[r] + b1r[nt]);
      }

      v16h aa[8];
#pragma unroll
      for (int kt = 0; kt < 8; ++kt)
        aa[kt] = load_afrag(actw + ln * ODE_H, kt * 32, g);

      // GEMM2: (16x256) x (256x128); h += dt * (acc + b2)
#pragma unroll
      for (int nt = 0; nt < 8; ++nt) {
        v8f c = {};
#pragma unroll
        for (int kt = 0; kt < 8; ++kt) {
          v16h b = *(const v16h*)(W2t + (nt * 8 + kt) * 512 + lane * 16);
          c = wmma_f16(aa[kt], b, c);
        }
        const int n = nt * 16 + ln;
#pragma unroll
        for (int r = 0; r < 8; ++r) {
          const int m = r + 8 * g;
          float hv = h32w[m * LAT + n] + dt * (c[r] + b2r[nt]);
          h32w[m * LAT + n] = hv;
          h16w[m * LAT + n] = (_Float16)hv;
        }
      }
    }

    // ---------------- GRU cell --------------------------------------------
    v16h ag[4];
#pragma unroll
    for (int kt = 0; kt < 4; ++kt)
      ag[kt] = load_afrag(h16w + ln * LAT, kt * 32, g);

#pragma unroll
    for (int nt = 0; nt < 8; ++nt) {
      v8f cr = {}, cz = {}, cn = {};
#pragma unroll
      for (int kt = 0; kt < 4; ++kt) {
        v16h br = *(const v16h*)(Wht + ((0  + nt) * 4 + kt) * 512 + lane * 16);
        v16h bz = *(const v16h*)(Wht + ((8  + nt) * 4 + kt) * 512 + lane * 16);
        v16h bn = *(const v16h*)(Wht + ((16 + nt) * 4 + kt) * 512 + lane * 16);
        cr = wmma_f16(ag[kt], br, cr);
        cz = wmma_f16(ag[kt], bz, cz);
        cn = wmma_f16(ag[kt], bn, cn);
      }
      const int n = nt * 16 + ln;
#pragma unroll
      for (int r = 0; r < 8; ++r) {
        const int m = r + 8 * g;
        const float x     = obsw[m];
        const float mk    = mskw[m];
        const float hprev = h32w[m * LAT + n];
        const float gr = fast_sigmoid(x * wir[nt] + bir[nt] + cr[r] + bhr[nt]);
        const float gz = fast_sigmoid(x * wiz[nt] + biz[nt] + cz[r] + bhz[nt]);
        const float gn = fast_tanh(x * win[nt] + bin[nt] + gr * (cn[r] + bhn[nt]));
        const float hc = (1.0f - gz) * gn + gz * hprev;
        const float hn2 = mk * hc + (1.0f - mk) * hprev;
        h32w[m * LAT + n] = hn2;
        h16w[m * LAT + n] = (_Float16)hn2;
      }
    }
  }

  // Emit final h (F x LAT, f32)
  for (int i = lane; i < 16 * LAT; i += 32) {
    const int fl = i / LAT;
    const int l  = i % LAT;
    out[(f0 + fl) * LAT + l] = h32w[i];
  }
}

extern "C" void kernel_launch(void* const* d_in, const int* in_sizes, int n_in,
                              void* d_out, int out_size, void* d_ws, size_t ws_size,
                              hipStream_t stream) {
  const float* times = (const float*)d_in[0];
  const float* vals  = (const float*)d_in[1];
  const float* mask  = (const float*)d_in[2];
  const float* W1    = (const float*)d_in[3];
  const float* b1    = (const float*)d_in[4];
  const float* W2    = (const float*)d_in[5];
  const float* b2    = (const float*)d_in[6];
  const float* W_ih  = (const float*)d_in[7];
  const float* b_ih  = (const float*)d_in[8];
  const float* Whh   = (const float*)d_in[9];
  const float* b_hh  = (const float*)d_in[10];

  _Float16* wf = (_Float16*)d_ws;
  float* out = (float*)d_out;

  prep_weights_kernel<<<(WHALFS + 255) / 256, 256, 0, stream>>>(W1, W2, Whh, wf);

  (void)hipFuncSetAttribute((const void*)ode_rnn_encoder_kernel,
                            hipFuncAttributeMaxDynamicSharedMemorySize,
                            LDS_TOTAL);

  const int blocks = NFEAT / (WAVES * 16);   // 64
  ode_rnn_encoder_kernel<<<blocks, WAVES * 32, LDS_TOTAL, stream>>>(
      times, vals, mask, b1, b2, W_ih, b_ih, b_hh, wf, out);
}
